// PDE_M2_10144712753408
// MI455X (gfx1250) — compile-verified, matrix operand order checked by reading, not run on hardware
//
#include <hip/hip_runtime.h>

#define N_MET   200000
#define N_RXN   400000
#define E_SUB   2000000
#define E_ALL   4000000
#define HIDDEN  64
#define MSG_DIM 32

typedef __attribute__((ext_vector_type(16))) _Float16 v16h;
typedef __attribute__((ext_vector_type(8)))  float    v8f;

// ---- fast transcendentals (gfx1250 V_TANH_F32 / V_EXP_F32 / V_LOG_F32) ----
__device__ __forceinline__ float fast_tanh(float x) {
#if __has_builtin(__builtin_amdgcn_tanhf)
  return __builtin_amdgcn_tanhf(x);
#else
  return tanhf(x);
#endif
}
__device__ __forceinline__ float fast_exp2(float x) {
#if __has_builtin(__builtin_amdgcn_exp2f)
  return __builtin_amdgcn_exp2f(x);
#else
  return exp2f(x);
#endif
}
__device__ __forceinline__ float fast_log2(float x) {
#if __has_builtin(__builtin_amdgcn_logf)
  return __builtin_amdgcn_logf(x);
#else
  return log2f(x);
#endif
}

// No-return f32 atomic add, SADDR (GVS) form: 64-bit uniform base in SGPRs,
// 32-bit per-lane byte offset in a VGPR -> zero 64-bit VALU address math.
__device__ __forceinline__ void atomic_add_f32_s(float* sbase, int voff_bytes, float v) {
  asm volatile("global_atomic_add_f32 %0, %1, %2"
               :: "v"(voff_bytes), "v"(v), "s"(sbase) : "memory");
}
__device__ __forceinline__ void atomic_add_f32_s64(float* sbase, int voff_bytes, float v) {
  asm volatile("global_atomic_add_f32 %0, %1, %2 offset:64"
               :: "v"(voff_bytes), "v"(v), "s"(sbase) : "memory");
}

// K-index mapping for 16-bit WMMA A/B operands (wave32, ISA 7.12.2):
// element i of a lane's v16h operand holds K = (i>>3)*16 + hi*8 + (i&7).
__device__ __forceinline__ int kmap(int i, int hi) {
  return ((i >> 3) << 4) + (hi << 3) + (i & 7);
}

// -------------------- kernel 0: zero accumulators --------------------
__global__ void k_zero(float* a, long long na, float* b, long long nb) {
  long long i = (long long)blockIdx.x * blockDim.x + threadIdx.x;
  long long stride = (long long)gridDim.x * blockDim.x;
  for (long long j = i; j < na; j += stride) a[j] = 0.0f;
  for (long long j = i; j < nb; j += stride) b[j] = 0.0f;
}

// -------------------- kernel 1: edge MLP (2->64 tanh ->32) + segment scatter --------------------
// One wave processes 16 edges/tile; layer 2 = 4x v_wmma_f32_16x16x32_f16.
// launch_bounds(.,1): lift the 128-VGPR occupancy cap so the ~190-reg working
// set (96 weight slices + 4 B operands + accumulators) stays resident (no spills).
__global__ void __launch_bounds__(256, 1)
k_edge_mlp(const float* __restrict__ x,
           const int*   __restrict__ met_sub,
           const int*   __restrict__ rxn_sub,
           const float* __restrict__ sto_sub,
           const float* __restrict__ W1,
           const float* __restrict__ b1,
           const float* __restrict__ W2,
           float* __restrict__ h_acc,
           float* __restrict__ ext_agg,
           float* __restrict__ cnt) {
  const int lane = threadIdx.x & 31;
  const int hi   = lane >> 4;
  const int l15  = lane & 15;

  // Loop-invariant per-lane W1/b1 slices; lane's 32 hidden indices follow the
  // A-operand layout: hidden(i) = (i>>3)*16 + hi*8 + (i&7), i<16 -> K-chunk 0.
  float w1c[32], w1s[32], b1v[32];
#pragma unroll
  for (int i = 0; i < 32; ++i) {
    int hidx = ((i >> 3) << 4) + (hi << 3) + (i & 7);   // 0..63
    w1c[i] = W1[hidx];            // W1[0][h]
    w1s[i] = W1[HIDDEN + hidx];   // W1[1][h]
    b1v[i] = b1[hidx];
  }
  // B operands = W2 (64x32): K-chunk c in {0,1} x N-half nh in {0,1}.
  v16h b00, b01, b10, b11;
#pragma unroll
  for (int i = 0; i < 16; ++i) {
    int kl = kmap(i, hi);
    b00[i] = (_Float16)W2[(kl     ) * MSG_DIM + (l15     )];
    b01[i] = (_Float16)W2[(kl     ) * MSG_DIM + (l15 + 16)];
    b10[i] = (_Float16)W2[(kl + 32) * MSG_DIM + (l15     )];
    b11[i] = (_Float16)W2[(kl + 32) * MSG_DIM + (l15 + 16)];
  }

  const int nTiles = E_SUB / 16;
  const int wavesPerBlock = blockDim.x >> 5;
  const int waveId = blockIdx.x * wavesPerBlock + (threadIdx.x >> 5);
  const int nWaves = gridDim.x * wavesPerBlock;

  for (int t = waveId; t < nTiles; t += nWaves) {
    const int base = t << 4;
    const int e = base + l15;          // lanes l and l+16 share edge l15
    const int m = met_sub[e];

    // Issue the scatter-index loads early so they overlap the MLP compute.
    int rx8[8];
#pragma unroll
    for (int r = 0; r < 8; ++r) rx8[r] = rxn_sub[base + hi * 8 + r];
    const int rx_lo = rxn_sub[e];      // for ext/cnt (lanes hi==0)

    const float conc = x[m * 8 + 3];
    const float sto  = sto_sub[e];

    // Layer 1: 32 hidden activations per lane, packed straight into A operands.
    v16h a0, a1;
#pragma unroll
    for (int i = 0; i < 32; ++i) {
      float h = fast_tanh(fmaf(conc, w1c[i], fmaf(sto, w1s[i], b1v[i])));
      if (i < 16) a0[i] = (_Float16)h; else a1[i - 16] = (_Float16)h;
    }

    // Layer 2: msg(16x32) = A(16x64) x W2(64x32), f32 accumulate.
    v8f acc0 = {}, acc1 = {};
    acc0 = __builtin_amdgcn_wmma_f32_16x16x32_f16(false, a0, false, b00, (short)0, acc0, false, false);
    acc0 = __builtin_amdgcn_wmma_f32_16x16x32_f16(false, a1, false, b10, (short)0, acc0, false, false);
    acc1 = __builtin_amdgcn_wmma_f32_16x16x32_f16(false, a0, false, b01, (short)0, acc1, false, false);
    acc1 = __builtin_amdgcn_wmma_f32_16x16x32_f16(false, a1, false, b11, (short)0, acc1, false, false);

    // D layout: VGPR r holds row M = hi*8 + r, column l15 (+16 for acc1).
    const int col_off = l15 * 4;
#pragma unroll
    for (int r = 0; r < 8; ++r) {
      const int off = rx8[r] * (MSG_DIM * 4) + col_off;   // max 51.2MB < 2^31
      atomic_add_f32_s  (h_acc, off, acc0[r]);
      atomic_add_f32_s64(h_acc, off, acc1[r]);            // +64B = column +16
    }
    // Scalar segment sums: one lane per edge.
    if (hi == 0) {
      atomic_add_f32_s(ext_agg, rx_lo * 4, x[m * 8 + 4]);
      atomic_add_f32_s(cnt,     rx_lo * 4, 1.0f);
    }
  }
}

// -------------------- kernel 2: reaction MLP (32->64 tanh ->1), rate law --------------------
__global__ void __launch_bounds__(256, 1)
k_rxn(const float* __restrict__ h_acc,
      const float* __restrict__ ext_agg,
      const float* __restrict__ cnt,
      const float* __restrict__ b2,
      const float* __restrict__ W3,
      const float* __restrict__ b3,
      const float* __restrict__ W4,
      const float* __restrict__ b4,
      const float* __restrict__ log_k,
      float* __restrict__ v_out) {
  const int lane = threadIdx.x & 31;
  const int hi   = lane >> 4;
  const int l15  = lane & 15;

  float b2v[16];
  v16h B0, B1, B2, B3;                 // W3 (32x64): 4 N-halves, single K=32 chunk
#pragma unroll
  for (int i = 0; i < 16; ++i) {
    int kl = kmap(i, hi);
    b2v[i] = b2[kl];
    B0[i] = (_Float16)W3[kl * HIDDEN + l15];
    B1[i] = (_Float16)W3[kl * HIDDEN + 16 + l15];
    B2[i] = (_Float16)W3[kl * HIDDEN + 32 + l15];
    B3[i] = (_Float16)W3[kl * HIDDEN + 48 + l15];
  }
  const float b3v0 = b3[l15], b3v1 = b3[16 + l15], b3v2 = b3[32 + l15], b3v3 = b3[48 + l15];
  const float w40 = W4[l15], w41 = W4[16 + l15], w42 = W4[32 + l15], w43 = W4[48 + l15];
  const float b4v = b4[0];

  const int nTiles = N_RXN / 16;
  const int wavesPerBlock = blockDim.x >> 5;
  const int waveId = blockIdx.x * wavesPerBlock + (threadIdx.x >> 5);
  const int nWaves = gridDim.x * wavesPerBlock;

  for (int t = waveId; t < nTiles; t += nWaves) {
    const int base = t << 4;
    const int rr = base + l15;         // A row = l15 for both lane halves
    const float c = cnt[rr];
    v16h a;
#pragma unroll
    for (int i = 0; i < 16; ++i) {
      int kl = kmap(i, hi);
      a[i] = (_Float16)(h_acc[(long long)rr * MSG_DIM + kl] + c * b2v[i]);  // fold cnt*b2
    }
    v8f acc0 = {}, acc1 = {}, acc2 = {}, acc3 = {};
    acc0 = __builtin_amdgcn_wmma_f32_16x16x32_f16(false, a, false, B0, (short)0, acc0, false, false);
    acc1 = __builtin_amdgcn_wmma_f32_16x16x32_f16(false, a, false, B1, (short)0, acc1, false, false);
    acc2 = __builtin_amdgcn_wmma_f32_16x16x32_f16(false, a, false, B2, (short)0, acc2, false, false);
    acc3 = __builtin_amdgcn_wmma_f32_16x16x32_f16(false, a, false, B3, (short)0, acc3, false, false);

    // tanh + W4 dot: lane holds 4 of 64 columns for rows M = hi*8 + r.
    float part[8];
#pragma unroll
    for (int r = 0; r < 8; ++r) {
      float s = fast_tanh(acc0[r] + b3v0) * w40;
      s = fmaf(fast_tanh(acc1[r] + b3v1), w41, s);
      s = fmaf(fast_tanh(acc2[r] + b3v2), w42, s);
      s = fmaf(fast_tanh(acc3[r] + b3v3), w43, s);
      part[r] = s;
    }
    // Butterfly reduce across 16 lanes of each half (masks 1..8 keep bit4 fixed).
#pragma unroll
    for (int r = 0; r < 8; ++r) {
#pragma unroll
      for (int mask = 1; mask <= 8; mask <<= 1)
        part[r] += __shfl_xor(part[r], mask, 32);
    }
    if (l15 == 0) {
#pragma unroll
      for (int r = 0; r < 8; ++r) {
        const int rx = base + hi * 8 + r;
        const float y = part[r] + b4v;
        float sp = (y > 20.0f) ? y
                 : 0.6931471805599453f * fast_log2(1.0f + fast_exp2(y * 1.4426950408889634f));
        const float k10 = fast_exp2(log_k[rx] * 3.3219280948873623f);  // 10^log_k
        const float denom = fmaxf(cnt[rx], 1.0f);
        v_out[rx] = 0.005f * k10 * (ext_agg[rx] / denom) * sp;          // 0.005 pre-folded
      }
    }
  }
}

// -------------------- kernel 3: final edge scatter into dxdt --------------------
__global__ void k_scatter(const int*   __restrict__ met_all,
                          const int*   __restrict__ rxn_all,
                          const float* __restrict__ sto_all,
                          const float* __restrict__ v,
                          float* __restrict__ out) {
  int i = blockIdx.x * blockDim.x + threadIdx.x;
  int stride = gridDim.x * blockDim.x;
  for (int e = i; e < E_ALL; e += stride) {
    atomic_add_f32_s(out, met_all[e] * 4, sto_all[e] * v[rxn_all[e]]);
  }
}

extern "C" void kernel_launch(void* const* d_in, const int* in_sizes, int n_in,
                              void* d_out, int out_size, void* d_ws, size_t ws_size,
                              hipStream_t stream) {
  const float* x       = (const float*)d_in[0];
  const int*   met_sub = (const int*)  d_in[1];
  const int*   rxn_sub = (const int*)  d_in[2];
  const float* sto_sub = (const float*)d_in[3];
  const int*   met_all = (const int*)  d_in[4];
  const int*   rxn_all = (const int*)  d_in[5];
  const float* sto_all = (const float*)d_in[6];
  const float* W1      = (const float*)d_in[7];
  const float* b1      = (const float*)d_in[8];
  const float* W2      = (const float*)d_in[9];
  const float* b2      = (const float*)d_in[10];
  const float* W3      = (const float*)d_in[11];
  const float* b3      = (const float*)d_in[12];
  const float* W4      = (const float*)d_in[13];
  const float* b4      = (const float*)d_in[14];
  const float* log_k   = (const float*)d_in[15];
  float* out = (float*)d_out;

  // Workspace (f32): h_acc[N_RXN*32] | ext_agg[N_RXN] | cnt[N_RXN] | v[N_RXN]  (~56 MB)
  float* h_acc   = (float*)d_ws;
  float* ext_agg = h_acc + (size_t)N_RXN * MSG_DIM;
  float* cnt     = ext_agg + N_RXN;
  float* v_arr   = cnt + N_RXN;

  const long long nz = (long long)N_RXN * MSG_DIM + 2LL * N_RXN;
  k_zero    <<<2048, 256, 0, stream>>>(h_acc, nz, out, (long long)N_MET);
  k_edge_mlp<<<1024, 256, 0, stream>>>(x, met_sub, rxn_sub, sto_sub, W1, b1, W2,
                                       h_acc, ext_agg, cnt);
  k_rxn     <<< 512, 256, 0, stream>>>(h_acc, ext_agg, cnt, b2, W3, b3, W4, b4, log_k, v_arr);
  k_scatter <<<2048, 256, 0, stream>>>(met_all, rxn_all, sto_all, v_arr, out);
}
